// MoELayer_36696200577223
// MI455X (gfx1250) — compile-verified, matrix operand order checked by reading, not run on hardware
//
#include <hip/hip_runtime.h>
#include <hip/hip_bf16.h>

// ---------------------------------------------------------------------------
// MoE (top-2 of 8 experts, SwiGLU FFN) for MI455X / gfx1250.
//   D=1024, F=4096, E=8, K=2, T=B*S=4096 tokens, fp32 in/out.
// bf16 WMMA GEMMs; B operands staged through LDS by the Tensor Data Mover
// (tensor_load_to_lds, double-buffered, s_wait_tensorcnt), shared by all 4
// waves of a block. Weights pre-converted to transposed bf16 (~201MB ~ L2).
// ---------------------------------------------------------------------------

#define D_DIM 1024
#define F_DIM 4096
#define E_NUM 8
#define T_TOK 4096          // B*S
#define R_TOT (T_TOK * 2)   // token-expert rows (top-2)

// K-slab staged per TDM transfer: 64 weight rows x 64 bf16 elements.
// LDS row = 32 dwords data + 4 dwords TDM pad = 144 B (bank-conflict spread).
#define KSLAB      64
#define ROW_BYTES  144
#define SLAB_BYTES (64 * ROW_BYTES)   // 9216 B
#define PAD_INTERVAL_CODE 4           // pad every 2^(4+1)=32 dwords (one row)
#define PAD_AMOUNT_CODE   3           // pad 3+1 = 4 dwords = 16 B

typedef __attribute__((ext_vector_type(16))) __bf16 v16bf;
typedef __attribute__((ext_vector_type(8)))  float  v8f;
typedef unsigned int uint32x4 __attribute__((ext_vector_type(4)));
typedef int          int32x4  __attribute__((ext_vector_type(4)));
typedef int          int32x8  __attribute__((ext_vector_type(8)));

#if __has_include(<hip/amd_detail/amd_gfx1250_TDM.h>)
#define TDM_6ARG 1
#else
#define TDM_6ARG 0
#endif

// 16-bit WMMA A/B per-lane data = two 16-byte chunks (ISA 7.12.2):
//   lanes 0-15 : K = [k0..k0+7]  and [k0+16..k0+23]
//   lanes 16-31: K = [k0+8..+15] and [k0+24..+31]   (k0 shifted by hi*8)
union ABFrag { v16bf v; uint4 q[2]; };

__device__ __forceinline__ unsigned short f32_to_bf16(float f) {
    unsigned int u = __float_as_uint(f);
    unsigned int r = u + 0x7FFFu + ((u >> 16) & 1u);   // round-nearest-even
    return (unsigned short)(r >> 16);
}

__device__ __forceinline__ void load_frag_g(ABFrag& f, const unsigned short* p) {
    const uint4* q = (const uint4*)p;   // 16B aligned by construction
    f.q[0] = q[0];
    f.q[1] = q[2];
}

__device__ __forceinline__ void load_frag_lds(ABFrag& f, const char* base,
                                              int row, int koff /*elems*/) {
    const char* p = base + row * ROW_BYTES + koff * 2;
    f.q[0] = *(const uint4*)p;          // ds_load_b128
    f.q[1] = *(const uint4*)(p + 32);
}

// ---------------------------------------------------------------------------
// TDM: async 2D tile load global -> LDS (D# per ISA 08_async_tensor §8).
//   tile: tileD1 rows x tileD0 bf16 elements, global row stride strideD0.
//   LDS: rows packed + 16B pad per row (pad_enable).
// ---------------------------------------------------------------------------
__device__ __forceinline__ void tdm_load_2d(unsigned int ldsOff,
                                            const void* gaddr,
                                            unsigned int tileD0,
                                            unsigned int tileD1,
                                            unsigned int strideD0) {
    unsigned long long ga = (unsigned long long)gaddr;
    const unsigned int bigDim = 1u << 20;   // tiles are always in-bounds
    uint32x4 g0;
    g0[0] = 1u;                                            // count=1, user D#
    g0[1] = ldsOff;                                        // lds_addr
    g0[2] = (unsigned int)(ga & 0xFFFFFFFFu);              // global_addr lo
    g0[3] = (unsigned int)((ga >> 32) & 0x01FFFFFFu) | (2u << 30); // type=2
    int32x8 g1;
    g1[0] = (int)((1u << 16) |                             // data_size = 2B
                  (1u << 20) |                             // pad_enable
                  ((unsigned)PAD_INTERVAL_CODE << 22) |
                  ((unsigned)PAD_AMOUNT_CODE << 25));
    g1[1] = (int)((bigDim & 0xFFFFu) << 16);               // tensor_dim0 lo16
    g1[2] = (int)(((bigDim >> 16) & 0xFFFFu) | ((bigDim & 0xFFFFu) << 16));
    g1[3] = (int)(((bigDim >> 16) & 0xFFFFu) | ((tileD0 & 0xFFFFu) << 16));
    g1[4] = (int)(tileD1 & 0xFFFFu);                       // tile_dim1, dim2=0
    g1[5] = (int)strideD0;                                 // dim0 stride lo32
    g1[6] = 0;                                             // stride hi, d1 stride
    g1[7] = 0;
    int32x4 z4 = {0, 0, 0, 0};
#if TDM_6ARG
    int32x8 z8 = {0, 0, 0, 0, 0, 0, 0, 0};
    __builtin_amdgcn_tensor_load_to_lds(g0, g1, z4, z4, z8, 0);
#else
    __builtin_amdgcn_tensor_load_to_lds(g0, g1, z4, z4, 0);
#endif
}

// ---------------------------------------------------------------------------
// Kernel 1: router (softmax fp32, top-2, counts). Writes router_probs.
// ---------------------------------------------------------------------------
__global__ void k_router(const float* __restrict__ x,
                         const float* __restrict__ gate_w,
                         float* __restrict__ probs_out,
                         int*   __restrict__ tokIdx,
                         float* __restrict__ tokW,
                         int*   __restrict__ ctrl) {
    int t = blockIdx.x * blockDim.x + threadIdx.x;
    if (t >= T_TOK) return;
    float logit[E_NUM];
#pragma unroll
    for (int e = 0; e < E_NUM; ++e) logit[e] = 0.f;
    const float* xr = x + (size_t)t * D_DIM;
    for (int d = 0; d < D_DIM; ++d) {
        float xv = xr[d];
        const float* g = gate_w + (size_t)d * E_NUM;
#pragma unroll
        for (int e = 0; e < E_NUM; ++e) logit[e] = fmaf(xv, g[e], logit[e]);
    }
    float m = logit[0];
#pragma unroll
    for (int e = 1; e < E_NUM; ++e) m = fmaxf(m, logit[e]);
    float p[E_NUM]; float s = 0.f;
#pragma unroll
    for (int e = 0; e < E_NUM; ++e) { p[e] = __expf(logit[e] - m); s += p[e]; }
    float inv = 1.f / s;
#pragma unroll
    for (int e = 0; e < E_NUM; ++e) {
        p[e] *= inv;
        probs_out[(size_t)t * E_NUM + e] = p[e];
    }
    int e1 = 0;
#pragma unroll
    for (int e = 1; e < E_NUM; ++e) if (p[e] > p[e1]) e1 = e;
    int e2 = (e1 == 0) ? 1 : 0;
#pragma unroll
    for (int e = 0; e < E_NUM; ++e) if (e != e1 && p[e] > p[e2]) e2 = e;
    float denom = p[e1] + p[e2];
    tokIdx[t * 2 + 0] = e1;  tokW[t * 2 + 0] = p[e1] / denom;
    tokIdx[t * 2 + 1] = e2;  tokW[t * 2 + 1] = p[e2] / denom;
    atomicAdd(&ctrl[e1], 1);
    atomicAdd(&ctrl[e2], 1);
}

// ---------------------------------------------------------------------------
// Kernel 2: exclusive prefix over 8 counts -> offsets; zero slot runners.
// ---------------------------------------------------------------------------
__global__ void k_offsets(int* __restrict__ ctrl) {
    if (threadIdx.x == 0) {
        int acc = 0;
#pragma unroll
        for (int e = 0; e < E_NUM; ++e) {
            int c = ctrl[e];
            ctrl[8 + e]  = acc;
            ctrl[16 + e] = 0;
            acc += c;
        }
    }
}

// ---------------------------------------------------------------------------
// Kernel 3: token-expert pair takes a slot, gathers x row as bf16.
// ---------------------------------------------------------------------------
__global__ void k_scatter(const float* __restrict__ x,
                          const int*   __restrict__ tokIdx,
                          const float* __restrict__ tokW,
                          int*   __restrict__ ctrl,
                          int*   __restrict__ rowTok,
                          float* __restrict__ rowW,
                          unsigned short* __restrict__ xg) {
    int pair = blockIdx.x;
    int t = pair >> 1;
    __shared__ int s_pos;
    if (threadIdx.x == 0) {
        int e = tokIdx[pair];
        int pos = ctrl[8 + e] + atomicAdd(&ctrl[16 + e], 1);
        rowTok[pos] = t;
        rowW[pos]   = tokW[pair];
        s_pos = pos;
    }
    __syncthreads();
    int pos = s_pos;
    const float* xr = x + (size_t)t * D_DIM;
    unsigned short* dst = xg + (size_t)pos * D_DIM;
    for (int d = threadIdx.x; d < D_DIM; d += blockDim.x)
        dst[d] = f32_to_bf16(xr[d]);
}

// ---------------------------------------------------------------------------
// Kernel 4: fp32 [Rr][Cc] -> bf16 [Cc][Rr] tiled transpose (weight staging).
// ---------------------------------------------------------------------------
__global__ void k_transpose_bf16(const float* __restrict__ src,
                                 unsigned short* __restrict__ dst,
                                 int Rr, int Cc) {
    __shared__ float tile[32][33];
    size_t bs = (size_t)Rr * Cc;
    const float* s = src + (size_t)blockIdx.z * bs;
    unsigned short* d = dst + (size_t)blockIdx.z * bs;
    int c0 = blockIdx.x * 32, r0 = blockIdx.y * 32;
#pragma unroll
    for (int k = 0; k < 4; ++k) {
        int r = r0 + threadIdx.y + k * 8;
        tile[threadIdx.y + k * 8][threadIdx.x] = s[(size_t)r * Cc + c0 + threadIdx.x];
    }
    __syncthreads();
#pragma unroll
    for (int k = 0; k < 4; ++k) {
        int c = c0 + threadIdx.y + k * 8;
        d[(size_t)c * Rr + r0 + threadIdx.x] = f32_to_bf16(tile[threadIdx.x][threadIdx.y + k * 8]);
    }
}

// ---------------------------------------------------------------------------
// Kernel 5: GEMM1 + SwiGLU. h[p,f] = silu(xg[p]·W1t[f]) * (xg[p]·W3t[f]).
// 4 waves x (16 rows x 64 cols); W1/W3 K-slabs TDM'd into LDS, double-buffered.
// No per-wave returns after the first barrier (ragged waves mask stores only).
// ---------------------------------------------------------------------------
__global__ __launch_bounds__(128)
void k_ffn1(const unsigned short* __restrict__ xg,   // [R][D]
            const unsigned short* __restrict__ w1t,  // [E][F][D]
            const unsigned short* __restrict__ w3t,  // [E][F][D]
            const int* __restrict__ ctrl,
            unsigned short* __restrict__ hbuf) {     // [R][F]
    __shared__ __align__(16) char sB[2][2][SLAB_BYTES];  // [buf][w1|w3]
    const int e   = blockIdx.z;
    const int cnt = ctrl[e];
    const int off = ctrl[8 + e];
    const int rowBlock = blockIdx.y * 64;
    if (rowBlock >= cnt) return;                     // block-uniform exit
    const int wave = threadIdx.x >> 5;
    const int lane = threadIdx.x & 31;
    const int hi = lane >> 4, ln = lane & 15;
    const int rowTile = rowBlock + wave * 16;
    const int colBase = blockIdx.x * 64;

    int aRow = rowTile + ln; if (aRow >= cnt) aRow = cnt - 1;   // clamp pads
    const unsigned short* aPtr = xg + (size_t)(off + aRow) * D_DIM;
    const size_t eW = (size_t)e * F_DIM * D_DIM;
    const unsigned short* w1base = w1t + eW + (size_t)colBase * D_DIM;
    const unsigned short* w3base = w3t + eW + (size_t)colBase * D_DIM;

    v8f acc1[4], acc3[4];
#pragma unroll
    for (int j = 0; j < 4; ++j) { acc1[j] = (v8f)0.f; acc3[j] = (v8f)0.f; }

    const int NS = D_DIM / KSLAB;                    // 16 slabs
    if (wave == 0) {
        tdm_load_2d((unsigned)(uintptr_t)&sB[0][0][0], w1base, KSLAB, 64, D_DIM);
        tdm_load_2d((unsigned)(uintptr_t)&sB[0][1][0], w3base, KSLAB, 64, D_DIM);
    }
    for (int s = 0; s < NS; ++s) {
        if (wave == 0) __builtin_amdgcn_s_wait_tensorcnt(0);
        __syncthreads();                             // slab s ready for all
        if (wave == 0 && s + 1 < NS) {               // prefetch next slab
            int nb = (s + 1) & 1;
            tdm_load_2d((unsigned)(uintptr_t)&sB[nb][0][0],
                        w1base + (size_t)(s + 1) * KSLAB, KSLAB, 64, D_DIM);
            tdm_load_2d((unsigned)(uintptr_t)&sB[nb][1][0],
                        w3base + (size_t)(s + 1) * KSLAB, KSLAB, 64, D_DIM);
        }
        const char* b1s = sB[s & 1][0];
        const char* b3s = sB[s & 1][1];
#pragma unroll
        for (int ks = 0; ks < KSLAB / 32; ++ks) {
            int kg = s * KSLAB + ks * 32;
            ABFrag a; load_frag_g(a, aPtr + kg + hi * 8);
#pragma unroll
            for (int j = 0; j < 4; ++j) {
                int r = j * 16 + ln;                 // tile-local weight row
                ABFrag b1, b3;
                load_frag_lds(b1, b1s, r, ks * 32 + hi * 8);
                load_frag_lds(b3, b3s, r, ks * 32 + hi * 8);
                acc1[j] = __builtin_amdgcn_wmma_f32_16x16x32_bf16(
                    false, a.v, false, b1.v, (short)0, acc1[j], false, false);
                acc3[j] = __builtin_amdgcn_wmma_f32_16x16x32_bf16(
                    false, a.v, false, b3.v, (short)0, acc3[j], false, false);
            }
        }
        __syncthreads();                             // done reading buf s&1
    }
    // C layout: VGPR i -> row i + 8*hi, col ln
#pragma unroll
    for (int j = 0; j < 4; ++j) {
#pragma unroll
        for (int i = 0; i < 8; ++i) {
            int rloc = rowTile + i + 8 * hi;
            if (rloc < cnt) {
                float c1 = acc1[j][i], c3 = acc3[j][i];
                float h = (c1 / (1.f + __expf(-c1))) * c3;
                hbuf[(size_t)(off + rloc) * F_DIM + colBase + j * 16 + ln] =
                    f32_to_bf16(h);
            }
        }
    }
}

// ---------------------------------------------------------------------------
// Kernel 6: GEMM2 + weighted scatter. out[tok[p],d] += w[p]*(h[p]·W2t[d]).
// Same TDM/LDS double-buffered B staging; 2 commutative fp32 atomics/element.
// ---------------------------------------------------------------------------
__global__ __launch_bounds__(128)
void k_ffn2(const unsigned short* __restrict__ hbuf, // [R][F]
            const unsigned short* __restrict__ w2t,  // [E][D][F]
            const int* __restrict__ ctrl,
            const int* __restrict__ rowTok,
            const float* __restrict__ rowW,
            float* __restrict__ out) {               // [T][D], pre-zeroed
    __shared__ __align__(16) char sB[2][SLAB_BYTES];
    __shared__ int   s_tok[64];
    __shared__ float s_w[64];
    const int e   = blockIdx.z;
    const int cnt = ctrl[e];
    const int off = ctrl[8 + e];
    const int rowBlock = blockIdx.y * 64;
    if (rowBlock >= cnt) return;                     // block-uniform exit
    if (threadIdx.x < 64) {
        int r = rowBlock + threadIdx.x;
        if (r >= cnt) r = cnt - 1;
        s_tok[threadIdx.x] = rowTok[off + r];
        s_w[threadIdx.x]   = rowW[off + r];
    }
    const int wave = threadIdx.x >> 5;
    const int lane = threadIdx.x & 31;
    const int hi = lane >> 4, ln = lane & 15;
    const int rowTile = rowBlock + wave * 16;
    const int colBase = blockIdx.x * 64;

    int aRow = rowTile + ln; if (aRow >= cnt) aRow = cnt - 1;
    const unsigned short* aPtr = hbuf + (size_t)(off + aRow) * F_DIM;
    const unsigned short* w2base =
        w2t + (size_t)e * D_DIM * F_DIM + (size_t)colBase * F_DIM;

    v8f acc[4];
#pragma unroll
    for (int j = 0; j < 4; ++j) acc[j] = (v8f)0.f;

    const int NS = F_DIM / KSLAB;                    // 64 slabs
    if (wave == 0)
        tdm_load_2d((unsigned)(uintptr_t)&sB[0][0], w2base, KSLAB, 64, F_DIM);
    for (int s = 0; s < NS; ++s) {
        if (wave == 0) __builtin_amdgcn_s_wait_tensorcnt(0);
        __syncthreads();
        if (wave == 0 && s + 1 < NS)
            tdm_load_2d((unsigned)(uintptr_t)&sB[(s + 1) & 1][0],
                        w2base + (size_t)(s + 1) * KSLAB, KSLAB, 64, F_DIM);
        const char* bs = sB[s & 1];
#pragma unroll
        for (int ks = 0; ks < KSLAB / 32; ++ks) {
            int kg = s * KSLAB + ks * 32;
            ABFrag a; load_frag_g(a, aPtr + kg + hi * 8);
#pragma unroll
            for (int j = 0; j < 4; ++j) {
                ABFrag b;
                load_frag_lds(b, bs, j * 16 + ln, ks * 32 + hi * 8);
                acc[j] = __builtin_amdgcn_wmma_f32_16x16x32_bf16(
                    false, a.v, false, b.v, (short)0, acc[j], false, false);
            }
        }
        __syncthreads();
    }
#pragma unroll
    for (int j = 0; j < 4; ++j) {
#pragma unroll
        for (int i = 0; i < 8; ++i) {
            int rloc = wave * 16 + i + 8 * hi;       // block-local row
            if (rowBlock + rloc < cnt) {
                float v = acc[j][i] * s_w[rloc];
                unsafeAtomicAdd(&out[(size_t)s_tok[rloc] * D_DIM +
                                     colBase + j * 16 + ln], v);
            }
        }
    }
}

// ---------------------------------------------------------------------------
// Host-side launch (graph-capture safe).
// ---------------------------------------------------------------------------
extern "C" void kernel_launch(void* const* d_in, const int* in_sizes, int n_in,
                              void* d_out, int out_size, void* d_ws, size_t ws_size,
                              hipStream_t stream) {
    (void)in_sizes; (void)n_in; (void)out_size; (void)ws_size;
    const float* x      = (const float*)d_in[0];
    const float* gate_w = (const float*)d_in[1];
    const float* w1     = (const float*)d_in[2];
    const float* w3     = (const float*)d_in[3];
    const float* w2     = (const float*)d_in[4];

    float* out   = (float*)d_out;                    // [T*D]
    float* probs = out + (size_t)T_TOK * D_DIM;      // [T*E]

    char* ws = (char*)d_ws;
    size_t o = 0;
    int*   ctrl   = (int*)(ws + o);           o += 256;
    int*   tokIdx = (int*)(ws + o);           o += (size_t)R_TOT * 4;
    float* tokW   = (float*)(ws + o);         o += (size_t)R_TOT * 4;
    int*   rowTok = (int*)(ws + o);           o += (size_t)R_TOT * 4;
    float* rowW   = (float*)(ws + o);         o += (size_t)R_TOT * 4;
    unsigned short* xg   = (unsigned short*)(ws + o); o += (size_t)R_TOT * D_DIM * 2;
    unsigned short* hbuf = (unsigned short*)(ws + o); o += (size_t)R_TOT * F_DIM * 2;
    unsigned short* w1t  = (unsigned short*)(ws + o); o += (size_t)E_NUM * F_DIM * D_DIM * 2;
    unsigned short* w3t  = (unsigned short*)(ws + o); o += (size_t)E_NUM * F_DIM * D_DIM * 2;
    unsigned short* w2t  = (unsigned short*)(ws + o); o += (size_t)E_NUM * D_DIM * F_DIM * 2;

    hipMemsetAsync(out, 0, (size_t)T_TOK * D_DIM * sizeof(float), stream);
    hipMemsetAsync(ctrl, 0, 256, stream);

    dim3 tb(32, 8, 1);
    k_transpose_bf16<<<dim3(F_DIM / 32, D_DIM / 32, E_NUM), tb, 0, stream>>>(w1, w1t, D_DIM, F_DIM);
    k_transpose_bf16<<<dim3(F_DIM / 32, D_DIM / 32, E_NUM), tb, 0, stream>>>(w3, w3t, D_DIM, F_DIM);
    k_transpose_bf16<<<dim3(D_DIM / 32, F_DIM / 32, E_NUM), tb, 0, stream>>>(w2, w2t, F_DIM, D_DIM);

    k_router<<<T_TOK / 256, 256, 0, stream>>>(x, gate_w, probs, tokIdx, tokW, ctrl);
    k_offsets<<<1, 32, 0, stream>>>(ctrl);
    k_scatter<<<R_TOT, 128, 0, stream>>>(x, tokIdx, tokW, ctrl, rowTok, rowW, xg);

    k_ffn1<<<dim3(F_DIM / 64, T_TOK / 64, E_NUM), 128, 0, stream>>>(xg, w1t, w3t, ctrl, hbuf);
    k_ffn2<<<dim3(D_DIM / 64, T_TOK / 64, E_NUM), 128, 0, stream>>>(hbuf, w2t, ctrl, rowTok, rowW, out);
}
